// PrithviSegBlowupT_12060268167764
// MI455X (gfx1250) — matrix operand order repaired and checked by reading, not compiled, not run on hardware
//
#include <hip/hip_runtime.h>
#include <hip/hip_bf16.h>

// ---------------------------------------------------------------------------
// PrithviSegBlowupT on gfx1250: WMMA f16 (f32 accum), double-buffered TDM,
// one-time f16 weight conversion so B fragments are two 16B loads.
// ---------------------------------------------------------------------------

typedef __attribute__((ext_vector_type(16))) _Float16 v16h;
typedef __attribute__((ext_vector_type(8)))  _Float16 h8;
typedef __attribute__((ext_vector_type(4)))  _Float16 h4;
typedef __attribute__((ext_vector_type(2)))  _Float16 h2;
typedef __attribute__((ext_vector_type(8)))  float    v8f;
typedef unsigned int u32x4 __attribute__((ext_vector_type(4)));
typedef int          i32x4 __attribute__((ext_vector_type(4)));
typedef int          i32x8 __attribute__((ext_vector_type(8)));

union fragU {
  v16h v;
  h8 h[2];
};

__device__ __forceinline__ v8f wmma_f16(v16h a, v16h b, v8f c) {
  return __builtin_amdgcn_wmma_f32_16x16x32_f16(false, a, false, b, (short)0, c,
                                                false, false);
}

// A fragment (16x32 f16) from row-major LDS: two contiguous 16B K-chunks.
__device__ __forceinline__ v16h loadA_lds(const _Float16* base, int ld,
                                          int koff) {
  const int lane = threadIdx.x & 31;
  const int m = lane & 15, kh = (lane >> 4) & 1;
  const _Float16* p = base + m * ld + koff + kh * 8;
  fragU f;
  f.h[0] = *(const h8*)p;
  f.h[1] = *(const h8*)(p + 16);
  return f.v;
}

// A fragment with K = 0..15 valid, 16..31 zero (upper chunk zeroed).
__device__ __forceinline__ v16h loadA_lds_k16(const _Float16* base, int ld) {
  const int lane = threadIdx.x & 31;
  const int m = lane & 15, kh = (lane >> 4) & 1;
  fragU f;
  f.h[0] = *(const h8*)(base + m * ld + kh * 8);
  f.h[1] = (h8)(_Float16)0;
  return f.v;
}

// B fragment (32x16 f16) from LDS: two contiguous 16B K-chunks per lane.
__device__ __forceinline__ v16h loadB_lds(const _Float16* base, int ld) {
  const int lane = threadIdx.x & 31;
  const int n = lane & 15, kh = (lane >> 4) & 1;
  const _Float16* p = base + n * ld + kh * 8;
  fragU f;
  f.h[0] = *(const h8*)p;
  f.h[1] = *(const h8*)(p + 16);
  return f.v;
}

// B fragment from a pre-converted global f16 weight matrix w[col][k]:
// two aligned 16B loads per lane, no conversion.
__device__ __forceinline__ v16h loadB_gf16(const _Float16* w, int ldk, int col,
                                           int kk) {
  const int lane = threadIdx.x & 31;
  const int n = lane & 15, kh = (lane >> 4) & 1;
  const _Float16* p = w + (col + n) * ldk + kk + kh * 8;
  fragU f;
  f.h[0] = *(const h8*)p;
  f.h[1] = *(const h8*)(p + 16);
  return f.v;
}

// Issue a 2-D TDM load: tile (tile0 x tile1) of f32, dim0 contiguous,
// dim1 stride = stride0 elements, into LDS at lds_byte_addr.
__device__ __forceinline__ void tdm_load_2d_f32(unsigned int lds_byte_addr,
                                                const float* gaddr,
                                                unsigned int tile0,
                                                unsigned int tile1,
                                                unsigned int stride0) {
  const unsigned long long ga = (unsigned long long)(uintptr_t)gaddr;
  u32x4 g0;
  g0[0] = 1u;                                   // count=1 valid descriptor
  g0[1] = lds_byte_addr;                        // lds_addr
  g0[2] = (unsigned int)ga;                     // global_addr[31:0]
  g0[3] = ((unsigned int)(ga >> 32) & 0x01FFFFFFu) | 0x80000000u;  // type=2
  i32x8 g1;
  g1[0] = (int)(2u << 16);                      // data_size = 4B
  g1[1] = (int)((tile0 & 0xFFFFu) << 16);       // tensor_dim0 = tile0
  g1[2] = (int)((tile0 >> 16) | ((tile1 & 0xFFFFu) << 16));  // tensor_dim1
  g1[3] = (int)((tile1 >> 16) | (tile0 << 16)); // tile_dim0
  g1[4] = (int)tile1;                           // tile_dim1 (tile_dim2 = 0)
  g1[5] = (int)stride0;                         // tensor_dim0_stride lo
  g1[6] = 0;
  g1[7] = 0;
  i32x4 gz = {0, 0, 0, 0};
#if defined(__clang_major__) && (__clang_major__ >= 23)
  i32x8 gz8 = {0, 0, 0, 0, 0, 0, 0, 0};
  __builtin_amdgcn_tensor_load_to_lds(g0, g1, gz, gz, gz8, 0);
#else
  __builtin_amdgcn_tensor_load_to_lds(g0, g1, gz, gz, 0);
#endif
}

// ---------------------------------------------------------------------------
// Kernel 0: one-time f32 -> f16 weight conversion (vectorized).
// ---------------------------------------------------------------------------
__global__ __launch_bounds__(256) void cvt_f32_to_f16_kernel(
    const float* __restrict__ in, _Float16* __restrict__ out, int nvec4) {
  const int i = blockIdx.x * 256 + threadIdx.x;
  if (i < nvec4) {
    const float4 v = ((const float4*)in)[i];
    h4 o;
    o[0] = (_Float16)v.x;
    o[1] = (_Float16)v.y;
    o[2] = (_Float16)v.z;
    o[3] = (_Float16)v.w;
    ((h4*)out)[i] = o;
  }
}

// ---------------------------------------------------------------------------
// Kernel 1: 1x1 conv GEMM (3072x1024 @ 1024x1024) + BN(eval) + exact GELU.
// Double-buffered TDM stages 16(M) x 128(K) f32 tiles; a branch-free repack
// converts each tile to WMMA A-fragment order; each wave owns 2 N-tiles so
// every A fragment feeds 2 WMMAs. Block = 16(M) x 128(N), 4 waves.
// ---------------------------------------------------------------------------
__global__ __launch_bounds__(128) void conv_bn_gelu_kernel(
    const float* __restrict__ x, const _Float16* __restrict__ cw16,
    const float* __restrict__ bn_g, const float* __restrict__ bn_b,
    const float* __restrict__ bn_m, const float* __restrict__ bn_v,
    float* __restrict__ g) {
  __shared__ __align__(16) float    aF32[2][128 * 16];   // [k][p] from TDM
  __shared__ __align__(32) _Float16 aFrag[2][4 * 512];   // 4 k-steps of frags

  const int tid = threadIdx.x;
  const int wv = tid >> 5, lane = tid & 31;
  const int mT = blockIdx.x >> 3;        // 0..191 row tile (frame t, 16 pix)
  const int nTg = blockIdx.x & 7;        // group of 8 N tiles (128 cols)
  const int t = mT >> 4;
  const int pbase = (mT & 15) << 4;
  const int obase = nTg * 128 + wv * 32;  // this wave: cols [obase, obase+32)

  const unsigned int ldsA0 = (unsigned int)(uintptr_t)&aF32[0][0];
  const unsigned int ldsA1 = (unsigned int)(uintptr_t)&aF32[1][0];

  // repack geometry: this thread owns k-pair (k0,k0+1) and 8 rows m0..m0+7
  const int k0 = (tid >> 1) << 1;
  const int m0 = (tid & 1) * 8;
  const int kk4 = k0 >> 5, kl = k0 & 31;
  const int kh = (kl >> 3) & 1;
  const int e = (kl & 7) + ((kl & 16) ? 8 : 0);

  if (wv == 0) {
    tdm_load_2d_f32(ldsA0, x + t * 256 + pbase, 16u, 128u, 3072u);
  }

  v8f c0 = {}, c1 = {};
  for (int kc = 0; kc < 8; ++kc) {
    const int buf = kc & 1;
    if (wv == 0) {
      if (kc < 7) {
        tdm_load_2d_f32(buf ? ldsA0 : ldsA1,
                        x + ((kc + 1) * 128 * 12 + t) * 256 + pbase, 16u, 128u,
                        3072u);
        __builtin_amdgcn_s_wait_tensorcnt(1);  // current chunk landed
      } else {
        __builtin_amdgcn_s_wait_tensorcnt(0);
      }
    }
    __syncthreads();
    // repack f32 [k][p] -> f16 fragment order (vector reads, b32 writes)
    {
      const float4* r0 = (const float4*)&aF32[buf][k0 * 16 + m0];
      const float4* r1 = (const float4*)&aF32[buf][(k0 + 1) * 16 + m0];
      const float4 a0 = r0[0], a1 = r0[1];
      const float4 b0 = r1[0], b1 = r1[1];
      const float va[8] = {a0.x, a0.y, a0.z, a0.w, a1.x, a1.y, a1.z, a1.w};
      const float vb[8] = {b0.x, b0.y, b0.z, b0.w, b1.x, b1.y, b1.z, b1.w};
      _Float16* fp = &aFrag[buf][kk4 * 512 + (kh * 16 + m0) * 16 + e];
#pragma unroll
      for (int m = 0; m < 8; ++m) {
        h2 pk;
        pk[0] = (_Float16)va[m];
        pk[1] = (_Float16)vb[m];
        *(h2*)(fp + m * 16) = pk;
      }
    }
    __syncthreads();
#pragma unroll
    for (int k4 = 0; k4 < 4; ++k4) {
      const v16h a = *(const v16h*)(&aFrag[buf][k4 * 512] + lane * 16);
      const int kk = kc * 128 + k4 * 32;
      c0 = wmma_f16(a, loadB_gf16(cw16, 1024, obase, kk), c0);
      c1 = wmma_f16(a, loadB_gf16(cw16, 1024, obase + 16, kk), c1);
    }
  }

  const int n = lane & 15, half = (lane >> 4) & 1;
#pragma unroll
  for (int tile = 0; tile < 2; ++tile) {
    const int o = obase + tile * 16 + n;
    const float ir = rsqrtf(bn_v[o] + 1e-5f);
    const float ga = bn_g[o], be = bn_b[o], mu = bn_m[o];
    const v8f& c = tile ? c1 : c0;
#pragma unroll
    for (int v = 0; v < 8; ++v) {
      const int m = v + half * 8;
      float val = (c[v] - mu) * ir * ga + be;
      val = 0.5f * val * (1.0f + erff(val * 0.70710678118654752f));
      g[(mT * 16 + m) * 1024 + o] = val;
    }
  }
}

// ---------------------------------------------------------------------------
// Kernel 2: per-pixel temporal transformer (one sequence per wave).
// ---------------------------------------------------------------------------
__device__ __forceinline__ void layer_norm_row(float* hS, _Float16* hH,
                                               const float* tmp,
                                               const float* gpt,
                                               const float* bpt, int lane) {
  const int row = lane & 15;  // both lane-halves duplicate rows (same writes)
  if (row < 12) {
    float xs[64];
    float mean = 0.f;
#pragma unroll
    for (int d = 0; d < 64; ++d) {
      const float xv = hS[row * 64 + d] + tmp[row * 64 + d];
      xs[d] = xv;
      mean += xv;
    }
    mean *= (1.f / 64.f);
    float var = 0.f;
#pragma unroll
    for (int d = 0; d < 64; ++d) {
      const float dd = xs[d] - mean;
      var += dd * dd;
    }
    var *= (1.f / 64.f);
    const float ir = rsqrtf(var + 1e-5f);
#pragma unroll
    for (int d = 0; d < 64; ++d) {
      const float yv = (xs[d] - mean) * ir * gpt[d] + bpt[d];
      hS[row * 64 + d] = yv;
      hH[row * 64 + d] = (_Float16)yv;
    }
  } else {
#pragma unroll
    for (int d = 0; d < 64; ++d) {
      hS[row * 64 + d] = 0.f;
      hH[row * 64 + d] = (_Float16)0;
    }
  }
}

__global__ __launch_bounds__(64) void transformer_kernel(
    const float* __restrict__ g, const float* __restrict__ w_in,
    const _Float16* __restrict__ qkv16, const float* __restrict__ qkv_b,
    const _Float16* __restrict__ ow16, const float* __restrict__ ob,
    const float* __restrict__ ln1_g, const float* __restrict__ ln1_b,
    const _Float16* __restrict__ ff116, const float* __restrict__ ff1_b,
    const _Float16* __restrict__ ff216, const float* __restrict__ ff2_b,
    const float* __restrict__ ln2_g, const float* __restrict__ ln2_b,
    const float* __restrict__ w_out, const float* __restrict__ b_out,
    float* __restrict__ out) {
  __shared__ __align__(32) float    hS[2][16 * 64];    // residual stream f32
  __shared__ __align__(32) _Float16 hH[2][16 * 64];    // f16 copy (WMMA A)
  __shared__ __align__(32) _Float16 sH[2][16 * 256];   // qkv / ff1 scratch
  __shared__ __align__(32) _Float16 oH[2][16 * 64];    // attn output
  __shared__ __align__(32) _Float16 aH[2][16 * 32];    // softmax, K-padded
  __shared__ __align__(32) _Float16 vT[2][64 * 32];    // V transposed [d][t]
  __shared__ __align__(32) float    tmpS[2][16 * 64];  // pre-LN accumulator

  const int wv = threadIdx.x >> 5;
  const int lane = threadIdx.x & 31;
  const int seq = blockIdx.x * 2 + wv;  // pixel index = hf*256 + wf

  __builtin_prefetch(qkv16, 0, 1);
  __builtin_prefetch(ff116, 0, 1);

  // zero padded regions once: aH cols 16..31, vT cols 16..31 (finite pads)
  for (int i = lane; i < 16 * 16; i += 32)
    aH[wv][(i >> 4) * 32 + 16 + (i & 15)] = (_Float16)0;
  for (int i = lane; i < 64 * 16; i += 32)
    vT[wv][(i >> 4) * 32 + 16 + (i & 15)] = (_Float16)0;

  // ---- gather pixel-shuffled conv activations, w_in projection + pos enc ---
  const int hf = seq >> 8, wf = seq & 255;
  const int p = ((hf >> 4) << 4) + (wf >> 4);
  const int obase = (hf & 15) * 16 + (wf & 15);
  const float* grow = g + p * 1024 + obase;
  for (int t = 0; t < 12; ++t) {
    const float* gt = grow + (t * 256) * 1024;
    const float v0 = gt[0], v1 = gt[256], v2 = gt[512], v3 = gt[768];
#pragma unroll
    for (int dd = 0; dd < 2; ++dd) {
      const int d = lane + dd * 32;
      float s = v0 * w_in[d * 4 + 0] + v1 * w_in[d * 4 + 1] +
                v2 * w_in[d * 4 + 2] + v3 * w_in[d * 4 + 3];
      const int i = d >> 1;
      const float divi = __expf((float)(2 * i) * (-9.2103403719762f / 64.0f));
      const float ang = (float)t * divi;
      s += (d & 1) ? __cosf(ang) : __sinf(ang);
      hS[wv][t * 64 + d] = s;
      hH[wv][t * 64 + d] = (_Float16)s;
    }
  }
  for (int i = lane; i < 4 * 64; i += 32) {  // zero pad rows 12..15
    hS[wv][12 * 64 + i] = 0.f;
    hH[wv][12 * 64 + i] = (_Float16)0;
  }
  __builtin_amdgcn_wave_barrier();

  const int n = lane & 15, half = (lane >> 4) & 1;

  for (int l = 0; l < 2; ++l) {
    // ---- QKV projection: (16x64) @ (64x192) -> sH f16 (+ vT for V) ---------
    for (int jt = 0; jt < 12; ++jt) {
      v8f c = {};
      for (int kk = 0; kk < 64; kk += 32) {
        const v16h a = loadA_lds(hH[wv], 64, kk);
        const v16h b = loadB_gf16(qkv16 + l * 192 * 64, 64, jt * 16, kk);
        c = wmma_f16(a, b, c);
      }
      const int col = jt * 16 + n;
      const float bias = qkv_b[l * 192 + col];
#pragma unroll
      for (int v = 0; v < 8; ++v) {
        const int m = v + half * 8;
        const _Float16 hv = (_Float16)(c[v] + bias);
        sH[wv][m * 256 + col] = hv;
        if (jt >= 8) vT[wv][(col - 128) * 32 + m] = hv;  // V transposed
      }
    }
    __builtin_amdgcn_wave_barrier();

    // ---- attention, one head per iteration ---------------------------------
    for (int hd = 0; hd < 4; ++hd) {
      // scores = q @ k^T  (K = HD = 16, upper A chunk zeroed)
      v8f c = {};
      {
        const v16h a = loadA_lds_k16(sH[wv] + hd * 16, 256);
        const v16h b = loadB_lds(sH[wv] + 64 + hd * 16, 256);
        c = wmma_f16(a, b, c);
      }
#pragma unroll
      for (int v = 0; v < 8; ++v) {
        float xv = (n < 12) ? c[v] * 0.25f : -1e30f;  // scale = 1/sqrt(16)
        float mx = xv;
        for (int msk = 1; msk < 16; msk <<= 1)
          mx = fmaxf(mx, __shfl_xor(mx, msk));
        const float e = (n < 12) ? __expf(xv - mx) : 0.f;
        float ssum = e;
        for (int msk = 1; msk < 16; msk <<= 1) ssum += __shfl_xor(ssum, msk);
        aH[wv][(v + half * 8) * 32 + n] = (_Float16)(e / ssum);
      }
      __builtin_amdgcn_wave_barrier();
      // o_head = att @ v  (K = T padded; att cols >= 12 are zero)
      v8f c2 = {};
      {
        const v16h a = loadA_lds(aH[wv], 32, 0);
        const v16h b = loadB_lds(vT[wv] + (hd * 16) * 32, 32);
        c2 = wmma_f16(a, b, c2);
      }
#pragma unroll
      for (int v = 0; v < 8; ++v)
        oH[wv][(v + half * 8) * 64 + hd * 16 + n] = (_Float16)c2[v];
      __builtin_amdgcn_wave_barrier();
    }

    // ---- output projection + residual + LN1 --------------------------------
    for (int jt = 0; jt < 4; ++jt) {
      v8f c = {};
      for (int kk = 0; kk < 64; kk += 32) {
        const v16h a = loadA_lds(oH[wv], 64, kk);
        const v16h b = loadB_gf16(ow16 + l * 64 * 64, 64, jt * 16, kk);
        c = wmma_f16(a, b, c);
      }
      const int col = jt * 16 + n;
      const float bias = ob[l * 64 + col];
#pragma unroll
      for (int v = 0; v < 8; ++v)
        tmpS[wv][(v + half * 8) * 64 + col] = c[v] + bias;
    }
    __builtin_amdgcn_wave_barrier();
    layer_norm_row(hS[wv], hH[wv], tmpS[wv], ln1_g + l * 64, ln1_b + l * 64,
                   lane);
    __builtin_amdgcn_wave_barrier();

    // ---- FF1: (16x64)@(64x256), relu -> sH ---------------------------------
    for (int jt = 0; jt < 16; ++jt) {
      v8f c = {};
      for (int kk = 0; kk < 64; kk += 32) {
        const v16h a = loadA_lds(hH[wv], 64, kk);
        const v16h b = loadB_gf16(ff116 + l * 256 * 64, 64, jt * 16, kk);
        c = wmma_f16(a, b, c);
      }
      const int col = jt * 16 + n;
      const float bias = ff1_b[l * 256 + col];
#pragma unroll
      for (int v = 0; v < 8; ++v)
        sH[wv][(v + half * 8) * 256 + col] = (_Float16)fmaxf(c[v] + bias, 0.f);
    }
    __builtin_amdgcn_wave_barrier();

    // ---- FF2: (16x256)@(256x64) + residual + LN2 ---------------------------
    for (int jt = 0; jt < 4; ++jt) {
      v8f c = {};
      for (int kk = 0; kk < 256; kk += 32) {
        const v16h a = loadA_lds(sH[wv], 256, kk);
        const v16h b = loadB_gf16(ff216 + l * 64 * 256, 256, jt * 16, kk);
        c = wmma_f16(a, b, c);
      }
      const int col = jt * 16 + n;
      const float bias = ff2_b[l * 64 + col];
#pragma unroll
      for (int v = 0; v < 8; ++v)
        tmpS[wv][(v + half * 8) * 64 + col] = c[v] + bias;
    }
    __builtin_amdgcn_wave_barrier();
    layer_norm_row(hS[wv], hH[wv], tmpS[wv], ln2_g + l * 64, ln2_b + l * 64,
                   lane);
    __builtin_amdgcn_wave_barrier();
  }

  // ---- head: mean over T, 1-class linear, sigmoid --------------------------
  float acc = 0.f;
#pragma unroll
  for (int dd = 0; dd < 2; ++dd) {
    const int d = lane + dd * 32;
    float s = 0.f;
    for (int t = 0; t < 12; ++t) s += hS[wv][t * 64 + d];
    acc += (s * (1.f / 12.f)) * w_out[d];
  }
  for (int msk = 1; msk < 32; msk <<= 1) acc += __shfl_xor(acc, msk);
  if (lane == 0) {
    const float logit = acc + b_out[0];
    out[seq] = 1.f / (1.f + __expf(-logit));
  }
}

// ---------------------------------------------------------------------------
extern "C" void kernel_launch(void* const* d_in, const int* in_sizes, int n_in,
                              void* d_out, int out_size, void* d_ws,
                              size_t ws_size, hipStream_t stream) {
  const float* x      = (const float*)d_in[0];
  const float* conv_w = (const float*)d_in[1];
  const float* bn_g   = (const float*)d_in[2];
  const float* bn_b   = (const float*)d_in[3];
  const float* bn_m   = (const float*)d_in[4];
  const float* bn_v   = (const float*)d_in[5];
  const float* w_in   = (const float*)d_in[6];
  const float* qkv_w  = (const float*)d_in[7];
  const float* qkv_b  = (const float*)d_in[8];
  const float* ow     = (const float*)d_in[9];
  const float* ob     = (const float*)d_in[10];
  const float* ln1_g  = (const float*)d_in[11];
  const float* ln1_b  = (const float*)d_in[12];
  const float* ff1_w  = (const float*)d_in[13];
  const float* ff1_b  = (const float*)d_in[14];
  const float* ff2_w  = (const float*)d_in[15];
  const float* ff2_b  = (const float*)d_in[16];
  const float* ln2_g  = (const float*)d_in[17];
  const float* ln2_b  = (const float*)d_in[18];
  const float* w_out  = (const float*)d_in[19];
  const float* b_out  = (const float*)d_in[20];
  float* out = (float*)d_out;

  // Workspace layout (256B-aligned offsets):
  char* ws = (char*)d_ws;
  float*    g     = (float*)(ws);                       // 3072*1024 f32
  _Float16* cw16  = (_Float16*)(ws + 12582912);         // 1024*1024 f16
  _Float16* qkv16 = (_Float16*)(ws + 12582912 + 2097152);          // 2*192*64
  _Float16* ow16  = (_Float16*)(ws + 12582912 + 2097152 + 49152);  // 2*64*64
  _Float16* ff116 = (_Float16*)(ws + 12582912 + 2097152 + 65536);  // 2*256*64
  _Float16* ff216 = (_Float16*)(ws + 12582912 + 2097152 + 131072); // 2*64*256

  // Stage 0: one-time weight conversion to f16.
  cvt_f32_to_f16_kernel<<<1024, 256, 0, stream>>>(conv_w, cw16, 1048576 / 4);
  cvt_f32_to_f16_kernel<<<24, 256, 0, stream>>>(qkv_w, qkv16, 24576 / 4);
  cvt_f32_to_f16_kernel<<<8, 256, 0, stream>>>(ow, ow16, 8192 / 4);
  cvt_f32_to_f16_kernel<<<32, 256, 0, stream>>>(ff1_w, ff116, 32768 / 4);
  cvt_f32_to_f16_kernel<<<32, 256, 0, stream>>>(ff2_w, ff216, 32768 / 4);

  // Stage 1: conv GEMM. Block = 16(M) x 128(N), 4 waves, TDM double-buffered.
  conv_bn_gelu_kernel<<<192 * 8, 128, 0, stream>>>(x, cw16, bn_g, bn_b, bn_m,
                                                   bn_v, g);
  // Stage 2: transformer, one sequence per wave, 2 waves/block.
  transformer_kernel<<<65536 / 2, 64, 0, stream>>>(
      g, w_in, qkv16, qkv_b, ow16, ob, ln1_g, ln1_b, ff116, ff1_b, ff216,
      ff2_b, ln2_g, ln2_b, w_out, b_out, out);
}
